// MultiHeadGraphAttentionLayer_52140902974318
// MI455X (gfx1250) — compile-verified, matrix operand order checked by reading, not run on hardware
//
#include <hip/hip_runtime.h>
#include <hip/hip_bf16.h>

// MI455X / gfx1250, wave32. Fused flash-attention GAT layer.
// f16 WMMA + f32 accum; TDM (tensor_load_to_lds) for K tiles,
// async global->LDS for V^T tiles, double-buffered.

#define B_  2
#define N_  4096
#define F_  256
#define H_  4
#define D_  64
#define KT_ 64                 // key-tile size
#define NT_ (N_ / KT_)         // 64 key tiles

typedef __attribute__((ext_vector_type(16))) _Float16 v16h;
typedef __attribute__((ext_vector_type(8)))  float    v8f;
typedef __attribute__((ext_vector_type(4)))  unsigned tdm4;
typedef __attribute__((ext_vector_type(8)))  unsigned tdm8;

union AFrag {
    v16h     v;
    _Float16 e[16];
    uint4    q[2];
};

static __device__ __forceinline__ v8f wmma_f16(const v16h& a, const v16h& b, const v8f& c) {
    return __builtin_amdgcn_wmma_f32_16x16x32_f16(false, a, false, b, (short)0, c, false, false);
}

// One 16-byte async global->LDS transfer per lane (ASYNCcnt-tracked).
static __device__ __forceinline__ void async_ld16(unsigned lds_off, const void* sbase,
                                                  unsigned voff) {
    asm volatile("global_load_async_to_lds_b128 %0, %1, %2"
                 :: "v"(lds_off), "v"(voff), "s"((unsigned long long)(uintptr_t)sbase)
                 : "memory");
}

// TDM: DMA one 2D f16 tile (tile_d0 x tile_d1) global -> LDS with row padding
// 128B data + 16B pad per row (matches the [..][72] LDS layout).
// 2-operand form => VADDR2/3 NULL => 2D descriptor. TENSORcnt-tracked.
static __device__ __forceinline__ void tdm_load_2d(unsigned lds_addr, const void* gaddr,
                                                   unsigned tensor_d0, unsigned tensor_d1,
                                                   unsigned stride0,
                                                   unsigned tile_d0, unsigned tile_d1) {
    const unsigned long long ga = (unsigned long long)(uintptr_t)gaddr;
    tdm4 g0;
    g0[0] = 1u;                                                 // count=1, load, user D#
    g0[1] = lds_addr;                                           // LDS byte address
    g0[2] = (unsigned)(ga & 0xFFFFFFFFu);                       // global_addr[31:0]
    g0[3] = (unsigned)((ga >> 32) & 0x01FFFFFFu) | (2u << 30);  // global_addr[56:32], type=2
    tdm8 g1;
    g1[0] = (1u << 16)                 // data_size = 2 bytes
          | (1u << 20)                 // pad_enable
          | (4u << 22)                 // pad_interval: 32 DWORDs (128B)
          | (3u << 25);                // pad_amount: 4 DWORDs (16B)
    g1[1] = (tensor_d0 & 0xFFFFu) << 16;                        // tensor_dim0[15:0]
    g1[2] = (tensor_d0 >> 16) | ((tensor_d1 & 0xFFFFu) << 16);  // dim0 hi | dim1 lo
    g1[3] = (tensor_d1 >> 16) | (tile_d0 << 16);                // dim1 hi | tile_dim0
    g1[4] = tile_d1;                                            // tile_dim1 (tile_dim2=0)
    g1[5] = stride0;                                            // tensor_dim0_stride[31:0]
    g1[6] = 0;                                                  // stride0 hi | stride1 lo
    g1[7] = 0;                                                  // stride1 hi
    asm volatile("tensor_load_to_lds %0, %1" :: "s"(g0), "s"(g1) : "memory");
}

// -------------------------------------------------------------------------
// Kernel 1: h_prime = h @ W per head, stored f16 twice:
//   hp  [H][B][N][64]  (row-major, Q/K fragments)
//   hpT [H][B][64][N]  (transposed, V^T tiles)
// -------------------------------------------------------------------------
__global__ __launch_bounds__(256) void proj_kernel(const float* __restrict__ hin,
                                                   const float* __restrict__ W,
                                                   _Float16* __restrict__ hp,
                                                   _Float16* __restrict__ hpT)
{
    __shared__ _Float16 Wt[64][264];   // W[head]^T, 528B rows (16B multiple)

    const int hb   = blockIdx.y;       // head*B + b
    const int head = hb / B_;
    const int b    = hb % B_;
    const int tid  = threadIdx.x;

    const float* Wh = W + (size_t)head * (F_ * D_);
    for (int i = tid; i < F_ * D_; i += 256) {
        int k = i >> 6, d = i & 63;
        Wt[d][k] = (_Float16)Wh[i];
    }
    __syncthreads();

    const int wave = tid >> 5;
    const int lane = tid & 31;
    const int m    = lane & 15;
    const int sel  = lane >> 4;
    const int rowbase = blockIdx.x * 128 + wave * 16;

    const float* hrow = hin + ((size_t)b * N_ + rowbase + m) * F_;

    v8f acc[4];
    const v8f vz = {0.f,0.f,0.f,0.f,0.f,0.f,0.f,0.f};
    #pragma unroll
    for (int c = 0; c < 4; c++) acc[c] = vz;

    #pragma unroll
    for (int kc = 0; kc < F_; kc += 32) {
        AFrag a;
        const int k0 = kc + sel * 8;
        #pragma unroll
        for (int i = 0; i < 8; i++) a.e[i]     = (_Float16)hrow[k0 + i];
        #pragma unroll
        for (int i = 0; i < 8; i++) a.e[8 + i] = (_Float16)hrow[k0 + 16 + i];

        #pragma unroll
        for (int c = 0; c < 4; c++) {
            AFrag bf;
            const _Float16* wt = &Wt[c * 16 + m][k0];
            bf.q[0] = *(const uint4*)(wt);
            bf.q[1] = *(const uint4*)(wt + 16);
            acc[c] = wmma_f16(a.v, bf.v, acc[c]);
        }
    }

    _Float16* hpb  = hp  + ((size_t)hb * N_ + rowbase) * D_;
    _Float16* hpTb = hpT + (size_t)hb * D_ * N_;
    #pragma unroll
    for (int c = 0; c < 4; c++)
        #pragma unroll
        for (int r = 0; r < 8; r++) {
            const int rr = r + 8 * sel;
            const int d  = c * 16 + m;
            const _Float16 v = (_Float16)acc[c][r];
            hpb[(size_t)rr * D_ + d] = v;
            hpTb[(size_t)d * N_ + rowbase + rr] = v;
        }
}

// -------------------------------------------------------------------------
// Kernel 2: flash attention per (head,b) + residual blend.
// 8 waves x 16 query rows; 64-key tiles; K tiles via TDM, V^T via async,
// double-buffered. Softmax denominator via 5th PV WMMA (ones column).
// -------------------------------------------------------------------------
__global__ __launch_bounds__(256) void attn_kernel(const _Float16* __restrict__ hp,
                                                   const _Float16* __restrict__ hpT,
                                                   const float* __restrict__ hin,
                                                   const float* __restrict__ alpha_p,
                                                   float* __restrict__ out)
{
    __shared__ _Float16 KV [2][64][72];   // keys row-major [k][d] (TDM, padded rows)
    __shared__ _Float16 KVT[2][80][72];   // V^T [d][k] (async); rows 64..79 ones/zeros
    __shared__ _Float16 Ptile[8][16][72]; // per-wave probabilities

    const int hb   = blockIdx.y;
    const int head = hb / B_;
    const int b    = hb % B_;
    const _Float16* hpb  = hp  + (size_t)hb * N_ * D_;   // [N][64]
    const _Float16* hpTb = hpT + (size_t)hb * D_ * N_;   // [64][N]

    const int tid  = threadIdx.x;
    const int wave = tid >> 5;
    const int lane = tid & 31;
    const int m    = lane & 15;
    const int sel  = lane >> 4;
    const int qbase = blockIdx.x * 128 + wave * 16;

    // ones/zeros rows of KVT (the l-accumulator V-tile), written once
    for (int i = tid; i < 2 * 16 * 72; i += 256) {
        const int bufi = i / (16 * 72);
        const int rem  = i % (16 * 72);
        const int rr   = rem / 72, cc = rem % 72;
        KVT[bufi][64 + rr][cc] = (rr == 0 && cc < 64) ? (_Float16)1.0f : (_Float16)0.0f;
    }

    // Q fragments in registers (two K-chunks over D=64)
    AFrag qf[2];
    const _Float16* qrow = hpb + (size_t)(qbase + m) * D_;
    #pragma unroll
    for (int kc2 = 0; kc2 < 2; kc2++) {
        const int k0 = kc2 * 32 + sel * 8;
        qf[kc2].q[0] = *(const uint4*)(qrow + k0);
        qf[kc2].q[1] = *(const uint4*)(qrow + k0 + 16);
    }

    float mrow[8];
    #pragma unroll
    for (int r = 0; r < 8; r++) mrow[r] = -1e30f;

    v8f o[5];
    const v8f vz = {0.f,0.f,0.f,0.f,0.f,0.f,0.f,0.f};
    #pragma unroll
    for (int c = 0; c < 5; c++) o[c] = vz;

    // Issue staging for key tile t into buffer buf:
    //  - K tile: one TDM descriptor (wave 0 only; TDM ignores EXEC)
    //  - V^T tile: 2 async b128 per lane
    auto issue_kv_tdm = [&](int t, int buf) {
        tdm_load_2d((unsigned)(uintptr_t)&KV[buf][0][0],
                    hpb + (size_t)t * KT_ * D_,
                    /*tensor_d0=*/D_, /*tensor_d1=*/N_, /*stride0=*/D_,
                    /*tile_d0=*/64, /*tile_d1=*/64);
    };
    auto issue_vt_async = [&](int t, int buf) {
        #pragma unroll
        for (int i = 0; i < 2; i++) {
            const int idx = tid * 2 + i;       // 512 granules of 16B per tile
            const int row = idx >> 3, cg = idx & 7;
            const unsigned ldsb = (unsigned)(uintptr_t)&KVT[buf][row][cg * 8];
            const unsigned vob  = (unsigned)((row * N_ + t * KT_ + cg * 8) * 2);
            async_ld16(ldsb, hpTb, vob);
        }
    };

    if (tid < 32) issue_kv_tdm(0, 0);
    issue_vt_async(0, 0);

    for (int t = 0; t < NT_; ++t) {
        const int buf = t & 1;
        if (t + 1 < NT_) {
            if (tid < 32) issue_kv_tdm(t + 1, buf ^ 1);
            issue_vt_async(t + 1, buf ^ 1);
            asm volatile("s_wait_asynccnt 2" ::: "memory");     // V^T tile t retired
            if (tid < 32) __builtin_amdgcn_s_wait_tensorcnt(1); // K tile t retired
        } else {
            asm volatile("s_wait_asynccnt 0" ::: "memory");
            if (tid < 32) __builtin_amdgcn_s_wait_tensorcnt(0);
        }
        __syncthreads();

        // ---- S = Q * K^T : four 16x16 halves, K=64 via 2 chained WMMA ----
        v8f s4[4];
        #pragma unroll
        for (int half = 0; half < 4; half++) {
            s4[half] = vz;
            #pragma unroll
            for (int kc2 = 0; kc2 < 2; kc2++) {
                AFrag bf;
                const _Float16* kr = &KV[buf][half * 16 + m][kc2 * 32 + sel * 8];
                bf.q[0] = *(const uint4*)(kr);
                bf.q[1] = *(const uint4*)(kr + 16);
                s4[half] = wmma_f16(qf[kc2].v, bf.v, s4[half]);
            }
        }

        // ---- LeakyReLU(0.2): max(v, 0.2v) ----
        #pragma unroll
        for (int half = 0; half < 4; half++)
            #pragma unroll
            for (int r = 0; r < 8; r++) {
                const float v = s4[half][r];
                s4[half][r] = fmaxf(v, 0.2f * v);
            }

        // ---- online softmax: only the row-MAX needs cross-lane reduction ----
        #pragma unroll
        for (int r = 0; r < 8; r++) {
            float tmax = fmaxf(fmaxf(s4[0][r], s4[1][r]), fmaxf(s4[2][r], s4[3][r]));
            #pragma unroll
            for (int off = 1; off < 16; off <<= 1)
                tmax = fmaxf(tmax, __shfl_xor(tmax, off, 32));
            const float mn    = fmaxf(mrow[r], tmax);
            const float scale = __expf(mrow[r] - mn);
            mrow[r] = mn;
            #pragma unroll
            for (int c = 0; c < 5; c++) o[c][r] *= scale;   // o[4] carries l
            #pragma unroll
            for (int half = 0; half < 4; half++)
                s4[half][r] = __expf(s4[half][r] - mn);
        }

        // ---- C-layout -> A-layout transpose of P through per-wave LDS ----
        _Float16 (*P)[72] = Ptile[wave];
        #pragma unroll
        for (int half = 0; half < 4; half++)
            #pragma unroll
            for (int r = 0; r < 8; r++)
                P[r + 8 * sel][half * 16 + m] = (_Float16)s4[half][r];

        AFrag pf[2];
        #pragma unroll
        for (int kc2 = 0; kc2 < 2; kc2++) {
            const _Float16* pr = &P[m][kc2 * 32 + sel * 8];
            pf[kc2].q[0] = *(const uint4*)(pr);
            pf[kc2].q[1] = *(const uint4*)(pr + 16);
        }

        // ---- O += P * V : 4 D-tiles + 1 l-tile (ones column) ----
        #pragma unroll
        for (int kc2 = 0; kc2 < 2; kc2++)
            #pragma unroll
            for (int c = 0; c < 5; c++) {
                AFrag vf;
                const _Float16* vr = &KVT[buf][c * 16 + m][kc2 * 32 + sel * 8];
                vf.q[0] = *(const uint4*)(vr);
                vf.q[1] = *(const uint4*)(vr + 16);
                o[c] = wmma_f16(pf[kc2].v, vf.v, o[c]);
            }
        __syncthreads();
    }

    // ---- epilogue: broadcast l from column 0 of the l-tile, blend, store ----
    float a = alpha_p[0];
    a = fminf(fmaxf(a, 0.f), 1.f);
    const float one_a = 1.f - a;
    float* outb = out + ((size_t)b * N_ + qbase) * F_;
    const float* hres = hin + ((size_t)b * N_ + qbase) * F_;
    #pragma unroll
    for (int r = 0; r < 8; r++) {
        const float l    = __shfl(o[4][r], lane & 16, 32);  // lane 0 / lane 16 holds l
        const float linv = 1.0f / l;
        const int   rr   = r + 8 * sel;
        #pragma unroll
        for (int c = 0; c < 4; c++) {
            const int col = head * 64 + c * 16 + m;
            const float val = o[c][r] * linv;
            outb[(size_t)rr * F_ + col] = a * val + one_a * hres[(size_t)rr * F_ + col];
        }
    }
}

extern "C" void kernel_launch(void* const* d_in, const int* in_sizes, int n_in,
                              void* d_out, int out_size, void* d_ws, size_t ws_size,
                              hipStream_t stream) {
    const float* h     = (const float*)d_in[0];   // [B,N,F] fp32
    const float* W     = (const float*)d_in[1];   // [H,F,D] fp32
    const float* alpha = (const float*)d_in[2];   // scalar fp32
    float* out = (float*)d_out;                   // [B,N,H*D] fp32

    _Float16* hp  = (_Float16*)d_ws;                          // 4 MB
    _Float16* hpT = hp + (size_t)H_ * B_ * N_ * D_;           // 4 MB

    dim3 grid(N_ / 128, H_ * B_);
    proj_kernel<<<grid, 256, 0, stream>>>(h, W, hp, hpT);
    attn_kernel<<<grid, 256, 0, stream>>>(hp, hpT, h, alpha, out);
}
